// SweetCocoaFilteredNoise_22574348108195
// MI455X (gfx1250) — matrix operand order repaired and checked by reading, not compiled, not run on hardware
//
#include <hip/hip_runtime.h>

typedef __attribute__((ext_vector_type(16))) _Float16 v16h;
typedef __attribute__((ext_vector_type(8)))  _Float16 v8h;
typedef __attribute__((ext_vector_type(8)))  float    v8f;
typedef __attribute__((ext_vector_type(4)))  int      v4i;

#define HOP    64      // frame hop / noise length
#define NB     64      // batch
#define NF     4000    // frames
#define NL     129     // H bins per frame
#define IRLEN  257     // linear-phase IR length (M = 2L-1)
#define KPAD   160     // K padded to 5 * 32 for wmma f16 k-steps
#define NPAD   272     // ir columns padded to 17 * 16
#define PADLO  68      // left zero pad of ir rows (covers m-j down to -68)
#define ROWW   400     // padded ir row: [0,68) zeros | 272 data | [340,400) zeros
#define SPAN   1280    // output span touched by 16 frames: 15*64 + 320
#define OUTLEN (NF * HOP)   // 256000 samples per batch row
#define GAIN   0.01f

#if __has_builtin(__builtin_amdgcn_global_load_async_to_lds_b128)
#define USE_ASYNC_LDS 1
typedef __attribute__((address_space(1))) v4i* gv4i_ptr;   // global AS1
typedef __attribute__((address_space(3))) v4i* lv4i_ptr;   // LDS    AS3
#endif

// ---------------------------------------------------------------------------
// Basis: C[k,n] = hann_periodic_257(n) * ((k==0?1:2)/257) * cos(2*pi*k*(n-128)/257)
// (irfft + roll(L-1) + hann window folded into one 129x257 cosine matrix,
//  zero-padded to 160x272 for WMMA tiling). f16 in d_ws (~87 KB, stays L2-hot).
// ---------------------------------------------------------------------------
__global__ __launch_bounds__(256) void build_basis(_Float16* __restrict__ bas) {
    int idx = blockIdx.x * 256 + threadIdx.x;
    if (idx >= KPAD * NPAD) return;
    int k = idx / NPAD;
    int n = idx % NPAD;
    float v = 0.0f;
    if (k < NL && n < IRLEN) {
        const float PI2 = 6.28318530717958647692f;
        float w     = 0.5f * (1.0f - __cosf(PI2 * (float)n / 257.0f));
        float scale = ((k == 0) ? 1.0f : 2.0f) / 257.0f;
        v = scale * w * __cosf(PI2 * (float)k * (float)(n - 128) / 257.0f);
    }
    bas[idx] = (_Float16)v;
}

// ---------------------------------------------------------------------------
// Fused kernel: one workgroup = 16 consecutive frames of one batch row.
//  Stage A: H tile -> LDS f16 (zero-padded K); noise tile -> LDS (async copy)
//  Stage B: ir_w = H x Basis via v_wmma_f32_16x16x32_f16, kept in LDS
//  Stage C: gather conv + OLA: 5 consecutive outputs per thread, rolling
//           register window over IR taps; plain stores for the exclusively
//           owned interior [256,1024), atomics only for the 512 edge samples.
// ---------------------------------------------------------------------------
__global__ __launch_bounds__(256) void fused_filtered_noise(
        const float*    __restrict__ H,
        const float*    __restrict__ noise,
        const _Float16* __restrict__ bas,
        float*          __restrict__ out) {

    __shared__ _Float16            Hlds[16][KPAD];   //  5.0 KB
    __shared__ float               irpad[16][ROWW];  // 25.6 KB zero-padded IRs
    __shared__ __align__(16) float nz[16][HOP];      //  4.0 KB

    const int tid  = threadIdx.x;
    const int lane = tid & 31;            // wave32
    const int wave = tid >> 5;            // 8 waves / workgroup
    const int f0   = blockIdx.x * 16;
    const int b    = blockIdx.y;

    // ---- Stage A: stage inputs into LDS -----------------------------------
    const float* Nrow = noise + ((size_t)b * NF + f0) * HOP;
#ifdef USE_ASYNC_LDS
    {   // raw 4 KB copy: one b128 per thread, tracked by ASYNCcnt
        gv4i_ptr gsrc = (gv4i_ptr)(Nrow + (size_t)tid * 4);
        lv4i_ptr ldst = (lv4i_ptr)(&nz[0][0] + tid * 4);
        __builtin_amdgcn_global_load_async_to_lds_b128(gsrc, ldst, 0, 0);
    }
#else
    for (int idx = tid; idx < 16 * HOP; idx += 256)
        nz[idx / HOP][idx % HOP] = Nrow[idx];
#endif

    const float* Hrow = H + ((size_t)b * NF + f0) * NL;
    for (int idx = tid; idx < 16 * KPAD; idx += 256) {
        int r = idx / KPAD, c = idx % KPAD;
        float v = (c < NL) ? Hrow[(size_t)r * NL + c] : 0.0f;
        Hlds[r][c] = (_Float16)v;
    }
    // zero-fill the pad columns of the IR rows (128 per row)
    for (int idx = tid; idx < 16 * 128; idx += 256) {
        int r = idx >> 7, c = idx & 127;
        irpad[r][(c < PADLO) ? c : (NPAD + c)] = 0.0f;   // [0,68) and [340,400)
    }
#ifdef USE_ASYNC_LDS
#if __has_builtin(__builtin_amdgcn_s_wait_asynccnt)
    __builtin_amdgcn_s_wait_asynccnt(0);
#else
    asm volatile("s_wait_asynccnt 0" ::: "memory");
#endif
#endif
    __syncthreads();

    // ---- Stage B: WMMA GEMM, D(16x16) tiles of ir_w -----------------------
    const int M  = lane & 15;   // A-matrix row held by this lane
    const int hi = lane >> 4;   // lane group (ISA 16-bit A layout)

    for (int nt = wave; nt < NPAD / 16; nt += 8) {     // 17 N-tiles over 8 waves
        const int n0 = nt * 16;
        v8f acc = {};
#pragma unroll
        for (int ks = 0; ks < KPAD / 32; ++ks) {       // 5 k-steps of 32
            const int kb = ks * 32;
            // A fragment: lane(M,hi): elems 0-7 = K kb+hi*8.., elems 8-15 = +16
            const int c0 = kb + hi * 8;
            v8h alo = *(const v8h*)&Hlds[M][c0];
            v8h ahi = *(const v8h*)&Hlds[M][c0 + 16];
            v16h a  = __builtin_shufflevector(alo, ahi,
                        0,1,2,3,4,5,6,7,8,9,10,11,12,13,14,15);
            // B fragment: lane holds column N = n0 + (lane&15), K in same order as A
            v16h bm;
#pragma unroll
            for (int s = 0; s < 16; ++s)
                bm[s] = bas[(size_t)(c0 + (s & 7) + (s >> 3) * 16) * NPAD + n0 + M];
            acc = __builtin_amdgcn_wmma_f32_16x16x32_f16(
                      false, a, false, bm, (short)0, acc, false, false);
        }
        // D layout: VGPR r -> row (r + hi*8), col n0 + (lane&15)
#pragma unroll
        for (int r = 0; r < 8; ++r)
            irpad[r + hi * 8][PADLO + n0 + M] = acc[r];
    }
    __syncthreads();

    // ---- Stage C: gather conv + overlap-add -------------------------------
    // Thread owns outputs rel = mstart .. mstart+4 (SPAN = 256*5).
    const int mstart = tid * 5;
    float acc0 = 0.f, acc1 = 0.f, acc2 = 0.f, acc3 = 0.f, acc4 = 0.f;

    for (int lf = 0; lf < 16; ++lf) {
        const int mb = mstart - 64 * lf;      // frame-local offset of rel u=0
        if (mb < -4 || mb > 319) continue;    // group has no overlap w/ frame
        const float* p = &irpad[lf][PADLO + mb];
        // rolling window w[u] = p[u - j]
        float w0 = p[0], w1 = p[1], w2 = p[2], w3 = p[3], w4 = p[4];
#pragma unroll
        for (int j = 0; j < HOP; ++j) {
            const float nzv = nz[lf][j];      // uniform -> LDS broadcast
            acc0 += nzv * w0;  acc1 += nzv * w1;  acc2 += nzv * w2;
            acc3 += nzv * w3;  acc4 += nzv * w4;
            if (j < HOP - 1) {                // shift window, 1 new tap / j
                w4 = w3; w3 = w2; w2 = w1; w1 = w0;
                w0 = p[-1 - j];
            }
        }
    }

    const size_t obase = (size_t)b * OUTLEN;
    const int    tbase = f0 * HOP;
    float accs[5] = {acc0, acc1, acc2, acc3, acc4};
#pragma unroll
    for (int u = 0; u < 5; ++u) {
        const int rel = mstart + u;
        const int t   = tbase + rel;
        if (t < OUTLEN) {
            const float v = accs[u] * GAIN;
            if (rel < 256 || rel >= 1024)      // shared with neighbor WG
                atomicAdd(out + obase + t, v);
            else                               // exclusively owned interior
                out[obase + t] = v;
        }
    }
}

// ---------------------------------------------------------------------------
extern "C" void kernel_launch(void* const* d_in, const int* in_sizes, int n_in,
                              void* d_out, int out_size, void* d_ws, size_t ws_size,
                              hipStream_t stream) {
    (void)in_sizes; (void)n_in; (void)ws_size;
    const float* H     = (const float*)d_in[0];   // (64, 4000, 129) f32
    const float* noise = (const float*)d_in[1];   // (64, 4000, 64)  f32
    float*       out   = (float*)d_out;           // (64, 256000)    f32
    _Float16*    bas   = (_Float16*)d_ws;         // 160x272 f16 basis (87 KB)

    // Edge regions accumulate with atomics -> zero output first (capturable).
    (void)hipMemsetAsync(d_out, 0, (size_t)out_size * sizeof(float), stream);

    build_basis<<<(KPAD * NPAD + 255) / 256, 256, 0, stream>>>(bas);

    dim3 grid(NF / 16, NB);   // 250 x 64 workgroups, 256 threads (8 waves) each
    fused_filtered_noise<<<grid, 256, 0, stream>>>(H, noise, bas, out);
}
